// PersonDetectorONNX_34368328302675
// MI455X (gfx1250) — compile-verified, matrix operand order checked by reading, not run on hardware
//
#include <hip/hip_runtime.h>
#include <math.h>

// ---------------- CDNA5 WMMA types ----------------
typedef __attribute__((ext_vector_type(16))) _Float16 v16h;
typedef __attribute__((ext_vector_type(8)))  float    v8f;

#define H_IN     720
#define W_IN     1280
#define NA       8400
#define MAXDET   150
#define THRESH   0.2f
#define IOU_THRV 0.7f
// resize_and_pad constants for 720x1280 -> 640: factor=0.5, th=360, tw=640,
// hph=140, hpw=0, x_factor=y_factor=2.0
#define HPH      140
#define CTR      319.5f   // (640-1)/2

__device__ __forceinline__ float sigmoidf(float x) { return 1.0f / (1.0f + __expf(-x)); }

// ---------------------------------------------------------------------------
// Kernel 0: per-image rotation index k from extrinsic matrix.
// cam_up = R[:, :3,:3] @ (0,-1,0)  =>  cam_up_i = -E[i][1]
// k = ((round(angle/(pi/2)) + 1) mod 4), round = RNE (rintf) like jnp.round.
// ---------------------------------------------------------------------------
__global__ void k_rot(const float* __restrict__ E, int B, int* __restrict__ kArr) {
  int b = blockIdx.x * blockDim.x + threadIdx.x;
  if (b >= B) return;
  float cx = -E[b * 16 + 0 * 4 + 1];
  float cy = -E[b * 16 + 1 * 4 + 1];
  float ang = atan2f(cy, cx);
  int kk = (int)rintf(ang * (float)(2.0 / M_PI)) + 1;
  kk = ((kk % 4) + 4) % 4;
  kArr[b] = kk;
}

// ---------------------------------------------------------------------------
// Kernel A: fused (2x2 area resize -> gamma 1/2.2 -> pad 0.5 -> rot90(k) ->
// stride-8 avg pool). Reads the 177MB image exactly once; writes only the
// [B,3,80,80] stride-8 feature map. Each thread produces one F8 cell by
// averaging the 8x8 block of rotated padded pixels; each padded pixel is the
// gamma of a 2x2 mean of original pixels (or the 0.5 pad constant).
// ---------------------------------------------------------------------------
__global__ void k_pool8(const float* __restrict__ img, const int* __restrict__ kArr,
                        float* __restrict__ F8, int B) {
  int idx = blockIdx.x * blockDim.x + threadIdx.x;
  int total = B * 3 * 80 * 80;
  if (idx >= total) return;
  int j = idx % 80;
  int i = (idx / 80) % 80;
  int c = (idx / 6400) % 3;
  int b = idx / 19200;
  int kk = kArr[b];
  const float* base = img + (size_t)(b * 3 + c) * (H_IN * W_IN);
  float acc = 0.0f;
  for (int dy = 0; dy < 8; ++dy) {
    int py = i * 8 + dy;
    for (int dx = 0; dx < 8; ++dx) {
      int px = j * 8 + dx;
      int qy, qx;  // map rotated coords -> padded-image coords (jnp.rot90 CCW)
      switch (kk) {
        case 0:  qy = py;        qx = px;        break;
        case 1:  qy = px;        qx = 639 - py;  break;
        case 2:  qy = 639 - py;  qx = 639 - px;  break;
        default: qy = 639 - px;  qx = py;        break;
      }
      float v;
      if (qy >= HPH && qy < HPH + 360) {
        const float* p = base + (size_t)(2 * (qy - HPH)) * W_IN + 2 * qx;
        float s2 = p[0] + p[1] + p[W_IN] + p[W_IN + 1];
        v = __powf(s2 * 0.25f, (1.0f / 2.2f));
      } else {
        v = 0.5f;  // pad constant
      }
      acc += v;
    }
  }
  F8[idx] = acc * (1.0f / 64.0f);
}

// ---------------------------------------------------------------------------
// Kernel B: stride-16 / stride-32 features = exact 2x2 / 4x4 averages of F8
// (pooling is linear, gamma already applied per-pixel before F8).
// ---------------------------------------------------------------------------
__global__ void k_down(const float* __restrict__ F8, float* __restrict__ F16s,
                       float* __restrict__ F32s, int B) {
  int idx = blockIdx.x * blockDim.x + threadIdx.x;
  int n16 = B * 3 * 40 * 40;
  int n32 = B * 3 * 20 * 20;
  if (idx < n16) {
    int x = idx % 40, y = (idx / 40) % 40, c = (idx / 1600) % 3, b = idx / 4800;
    const float* f = F8 + (size_t)(b * 3 + c) * 6400;
    float s = f[(2 * y) * 80 + 2 * x] + f[(2 * y) * 80 + 2 * x + 1] +
              f[(2 * y + 1) * 80 + 2 * x] + f[(2 * y + 1) * 80 + 2 * x + 1];
    F16s[idx] = s * 0.25f;
  } else if (idx < n16 + n32) {
    int m = idx - n16;
    int x = m % 20, y = (m / 20) % 20, c = (m / 400) % 3, b = m / 1200;
    const float* f = F8 + (size_t)(b * 3 + c) * 6400;
    float s = 0.0f;
    for (int dy = 0; dy < 4; ++dy)
      for (int dx = 0; dx < 4; ++dx)
        s += f[(4 * y + dy) * 80 + 4 * x + dx];
    F32s[m] = s * (1.0f / 16.0f);
  }
}

// ---------------------------------------------------------------------------
// Kernel C: head projection on the matrix unit + YOLOX decode.
// Per wave: D = A(16x32 f16) x B(32x16 f16) + 0 via v_wmma_f32_16x16x32_f16,
// with A rows = 16 anchors' features (K=0..2, rest zero) and B = head_w
// (K=0..2 rows, N=0..5 cols, rest zero). ISA lane layouts (CDNA5 ISA):
//   A, lane<16:  M=lane,    v16h elems 0..7 = K 0..7      (we fill K=0..2)
//   A, lane>=16: M=lane-16, elems = K 8..15 / 24..31       (all zero here)
//   B, lane<16:  N=lane,    elems 0..7 = K 0..7          (we fill K=0..2)
//   B, lane>=16: N=lane-16, elems = K 8..15 / 24..31               (all zero here)
//   D, lane<16:  elem r = D[M=r][N=lane]; lane>=16: D[M=8+r][N=lane-16]
// ---------------------------------------------------------------------------
__global__ void k_decode(const float* __restrict__ F8, const float* __restrict__ F16s,
                         const float* __restrict__ F32s, const float* __restrict__ raw_preds,
                         const float* __restrict__ head_w, float4* __restrict__ boxes,
                         float* __restrict__ scores, int B) {
  __shared__ float tileLds[8][16 * 6];
  const int wave = threadIdx.x >> 5;  // wave32
  const int lane = threadIdx.x & 31;
  const int blocksPerImg = 66;        // 66 blocks * 8 waves * 16 anchors >= 8400
  int b = blockIdx.x / blocksPerImg;
  int tile = (blockIdx.x % blocksPerImg) * 8 + wave;
  int anchor0 = tile * 16;

  // A fragment: per-anchor features (K padded 3 -> 32 with zeros)
  v16h a;
#pragma unroll
  for (int e = 0; e < 16; ++e) a[e] = (_Float16)0.0f;
  if (lane < 16) {
    int anchor = anchor0 + lane;
    if (anchor >= NA) anchor = NA - 1;  // clamp; tail stores are guarded below
    float f0, f1, f2;
    if (anchor < 6400) {
      const float* p = F8 + (size_t)b * 3 * 6400;
      f0 = p[anchor]; f1 = p[6400 + anchor]; f2 = p[12800 + anchor];
    } else if (anchor < 8000) {
      int m = anchor - 6400;
      const float* p = F16s + (size_t)b * 3 * 1600;
      f0 = p[m]; f1 = p[1600 + m]; f2 = p[3200 + m];
    } else {
      int m = anchor - 8000;
      const float* p = F32s + (size_t)b * 3 * 400;
      f0 = p[m]; f1 = p[400 + m]; f2 = p[800 + m];
    }
    a[0] = (_Float16)f0; a[1] = (_Float16)f1; a[2] = (_Float16)f2;
  }

  // B fragment: head_w [3,6] row-major
  v16h bw;
#pragma unroll
  for (int e = 0; e < 16; ++e) bw[e] = (_Float16)0.0f;
  if (lane < 6) {
    bw[0] = (_Float16)head_w[0 * 6 + lane];
    bw[1] = (_Float16)head_w[1 * 6 + lane];
    bw[2] = (_Float16)head_w[2 * 6 + lane];
  }

  v8f c = {};
  // 8 args: (neg_a, A, neg_b, B, c_mod, C, reuse_a, reuse_b)
  c = __builtin_amdgcn_wmma_f32_16x16x32_f16(false, a, false, bw, (short)0, c, false, false);

  // scatter D tile to LDS so each anchor's row is contiguous for decode
  {
    int n  = (lane < 16) ? lane : lane - 16;
    int mb = (lane < 16) ? 0 : 8;
    if (n < 6) {
#pragma unroll
      for (int r = 0; r < 8; ++r) tileLds[wave][(mb + r) * 6 + n] = c[r];
    }
  }
  __syncthreads();

  if (lane < 16) {
    int anchor = anchor0 + lane;
    if (anchor < NA) {
      const float* rp = raw_preds + ((size_t)b * NA + anchor) * 6;
      float raw[6];
#pragma unroll
      for (int jj = 0; jj < 6; ++jj) raw[jj] = rp[jj] + tileLds[wave][lane * 6 + jj];
      int s, gx, gy;
      if (anchor < 6400)      { s = 8;  gx = anchor % 80; gy = anchor / 80; }
      else if (anchor < 8000) { int m = anchor - 6400; s = 16; gx = m % 40; gy = m / 40; }
      else                    { int m = anchor - 8000; s = 32; gx = m % 20; gy = m / 20; }
      float fs = (float)s;
      float cxx = (raw[0] + (float)gx) * fs;
      float cyy = (raw[1] + (float)gy) * fs;
      float ww  = __expf(raw[2]) * fs;
      float hh  = __expf(raw[3]) * fs;
      boxes[(size_t)b * NA + anchor] =
          make_float4(cxx - 0.5f * ww, cyy - 0.5f * hh, cxx + 0.5f * ww, cyy + 0.5f * hh);
      scores[(size_t)b * NA + anchor] = sigmoidf(raw[4]) * sigmoidf(raw[5]);
    }
  }
}

// ---------------------------------------------------------------------------
// Kernel D: greedy NMS entirely resident in CDNA5's 320KB LDS (171KB used:
// 8400 float4 boxes + 8400 scores + reduction scratch), one workgroup per
// image, fixed MAX_DET iterations (torchvision semantics), then the
// rotation-undo + rescale of scale_boxes_single, writing [B,150,5].
// ---------------------------------------------------------------------------
__global__ void k_nms(const float4* __restrict__ boxes, const float* __restrict__ scores,
                      const int* __restrict__ kArr, float* __restrict__ out, int B) {
  extern __shared__ char smem[];
  float4* sB  = (float4*)smem;             // 8400 * 16 B
  float*  sS  = (float*)(sB + NA);         // 8400 * 4 B
  float*  rV  = sS + NA;                   // 256  * 4 B
  int*    rI  = (int*)(rV + 256);          // 256  * 4 B
  float*  selS = (float*)(rI + 256);       // 160  * 4 B
  int*    selI = (int*)(selS + 160);       // 160  * 4 B

  int b = blockIdx.x;
  int tid = threadIdx.x;

  for (int n = tid; n < NA; n += 256) {
    float4 bb = boxes[(size_t)b * NA + n];
    sB[n] = bb;
    float sc = scores[(size_t)b * NA + n];
    sS[n] = (sc > THRESH) ? sc : -INFINITY;
  }
  __syncthreads();

  for (int it = 0; it < MAXDET; ++it) {
    // argmax with first-index tie-break (jnp.argmax semantics)
    float bv = -INFINITY;
    int   bi = 0x7fffffff;
    for (int n = tid; n < NA; n += 256) {
      float v = sS[n];
      if (v > bv || (v == bv && n < bi)) { bv = v; bi = n; }
    }
    rV[tid] = bv; rI[tid] = bi;
    __syncthreads();
    for (int off = 128; off > 0; off >>= 1) {
      if (tid < off) {
        float v = rV[tid + off]; int i2 = rI[tid + off];
        if (v > rV[tid] || (v == rV[tid] && i2 < rI[tid])) { rV[tid] = v; rI[tid] = i2; }
      }
      __syncthreads();
    }
    int i = rI[0];
    float sc = rV[0];
    if (tid == 0) { selS[it] = sc; selI[it] = i; }
    float4 bb = sB[i];
    float  ba = (bb.z - bb.x) * (bb.w - bb.y);
    __syncthreads();  // everyone has read rV/rI[0] before they are reused

    for (int n = tid; n < NA; n += 256) {
      float4 cb = sB[n];
      float ltx = fmaxf(bb.x, cb.x), lty = fmaxf(bb.y, cb.y);
      float rbx = fminf(bb.z, cb.z), rby = fminf(bb.w, cb.w);
      float iw = fmaxf(rbx - ltx, 0.0f), ih = fmaxf(rby - lty, 0.0f);
      float inter = iw * ih;
      float area_n = (cb.z - cb.x) * (cb.w - cb.y);
      float iou = inter / (area_n + ba - inter + 1e-9f);
      if (iou > IOU_THRV) sS[n] = -INFINITY;
    }
    if (tid == 0) sS[i] = -INFINITY;
    __syncthreads();
  }

  // scale_boxes_single: undo rot90(k), rescale to original resolution
  int kk = kArr[b];
  float ck = (kk == 0) ? 1.0f : ((kk == 2) ? -1.0f : 0.0f);
  float sk = (kk == 1) ? 1.0f : ((kk == 3) ? -1.0f : 0.0f);
  for (int d = tid; d < MAXDET; d += 256) {
    float* o = out + ((size_t)b * MAXDET + d) * 5;
    float sc = selS[d];
    if (sc > -INFINITY) {
      float4 bb = sB[selI[d]];
      float mx = (bb.x + bb.z) * 0.5f - CTR;
      float my = (bb.y + bb.w) * 0.5f - CTR;
      float rx = ck * mx - sk * my + CTR;
      float ry = sk * mx + ck * my + CTR;
      float sx = bb.z - bb.x, sy = bb.w - bb.y;
      if (kk & 1) { float t = sx; sx = sy; sy = t; }
      float tlx = rx - 0.5f * sx;
      float tly = ry - 0.5f * sy;
      o[0] = tlx * 2.0f;                 // hpw = 0, x_factor = 2
      o[1] = (tly - (float)HPH) * 2.0f;  // hph = 140, y_factor = 2
      o[2] = sx * 2.0f;
      o[3] = sy * 2.0f;
      o[4] = sc;
    } else {
      o[0] = 0.0f; o[1] = 0.0f; o[2] = 0.0f; o[3] = 0.0f; o[4] = 0.0f;
    }
  }
}

// ---------------------------------------------------------------------------
extern "C" void kernel_launch(void* const* d_in, const int* in_sizes, int n_in,
                              void* d_out, int out_size, void* d_ws, size_t ws_size,
                              hipStream_t stream) {
  (void)n_in; (void)out_size; (void)ws_size;
  const float* images    = (const float*)d_in[0];  // [B,3,720,1280]
  const float* raw_preds = (const float*)d_in[1];  // [B,8400,6]
  const float* E         = (const float*)d_in[2];  // [B,4,4]
  const float* head_w    = (const float*)d_in[3];  // [3,6]
  int B = in_sizes[0] / (3 * H_IN * W_IN);

  // workspace layout (all 256B aligned), total ~4.2 MB for B=16
  char* ws = (char*)d_ws;
  size_t off = 0;
  int*    kArr = (int*)(ws + off);   off += 256;
  float*  F8   = (float*)(ws + off); off += (size_t)B * 3 * 6400 * 4;
  float*  F16s = (float*)(ws + off); off += (size_t)B * 3 * 1600 * 4;
  float*  F32s = (float*)(ws + off); off += (size_t)B * 3 * 400 * 4;
  off = (off + 255) & ~(size_t)255;
  float4* boxesW  = (float4*)(ws + off); off += (size_t)B * NA * 16;
  float*  scoresW = (float*)(ws + off);  off += (size_t)B * NA * 4;

  k_rot<<<1, 32, 0, stream>>>(E, B, kArr);

  int totalA = B * 3 * 80 * 80;
  k_pool8<<<(totalA + 255) / 256, 256, 0, stream>>>(images, kArr, F8, B);

  int totalB = B * 3 * (1600 + 400);
  k_down<<<(totalB + 255) / 256, 256, 0, stream>>>(F8, F16s, F32s, B);

  k_decode<<<B * 66, 256, 0, stream>>>(F8, F16s, F32s, raw_preds, head_w, boxesW, scoresW, B);

  size_t smem = (size_t)NA * 16 + (size_t)NA * 4 + 256 * 4 + 256 * 4 + 160 * 4 + 160 * 4;
  k_nms<<<B, 256, smem, stream>>>(boxesW, scoresW, kArr, (float*)d_out, B);
}
